// MultiHeadAttention_2259152797992
// MI455X (gfx1250) — matrix operand order repaired and checked by reading, not compiled, not run on hardware
//
#include <hip/hip_runtime.h>
#include <hip/hip_bf16.h>

// Problem constants (from reference): B=2, T=2048, C=1024, H=16, D=64
#define BB 2
#define TT 2048
#define CC 1024
#define HH 16
#define DD 64

typedef _Float16 v16h __attribute__((ext_vector_type(16)));
typedef _Float16 v8h  __attribute__((ext_vector_type(8)));
typedef float    v8f  __attribute__((ext_vector_type(8)));

static __device__ __forceinline__ v16h combine16(v8h lo, v8h hi) {
  return __builtin_shufflevector(lo, hi, 0,1,2,3,4,5,6,7,8,9,10,11,12,13,14,15);
}

// A-fragment: 16x32 f16, row-major source base[m*ld + k].
// lane<16: M=lane, V0-3 = K k0+0..7, V4-7 = K k0+16..23
// lane>=16: M=lane-16, V0-3 = K k0+8..15, V4-7 = K k0+24..31
static __device__ __forceinline__ v16h load_a_frag(const _Float16* base, int ld, int lane, int k0) {
  int m = lane & 15;
  int koff = (lane >> 4) << 3;
  const _Float16* p = base + (size_t)m * ld + k0 + koff;
  v8h lo = *reinterpret_cast<const v8h*>(p);
  v8h hi = *reinterpret_cast<const v8h*>(p + 16);
  return combine16(lo, hi);
}

// B-fragment: 32x16 f16, from transposed storage baseT[n*ld + k] == B[k][n].
// lane<16: N=lane, K = k0+0..7, k0+16..23 ; lane>=16: N=lane-16, K = k0+8..15, k0+24..31
static __device__ __forceinline__ v16h load_b_frag(const _Float16* baseT, int ld, int lane, int k0) {
  int n = lane & 15;
  int kb = k0 + ((lane >> 4) << 3);
  const _Float16* p = baseT + (size_t)n * ld + kb;
  v8h lo = *reinterpret_cast<const v8h*>(p);
  v8h hi = *reinterpret_cast<const v8h*>(p + 16);
  return combine16(lo, hi);
}

// ---------------------------------------------------------------- prep kernels

__global__ void conv_x_kernel(const float* __restrict__ x, _Float16* __restrict__ xh, int n) {
  int i = blockIdx.x * blockDim.x + threadIdx.x;
  if (i < n) xh[i] = (_Float16)x[i];
}

// wt[((m*H+h)*D + d)*C + c] = W_m[(h*C + c)*D + d]   (transpose C<->D per head)
__global__ void conv_w_kernel(const float* __restrict__ wq, const float* __restrict__ wk,
                              const float* __restrict__ wv, _Float16* __restrict__ wt) {
  int i = blockIdx.x * blockDim.x + threadIdx.x;
  const int ntot = 3 * HH * DD * CC;
  if (i >= ntot) return;
  int c = i % CC;
  int tmp = i / CC;
  int d = tmp % DD;
  tmp /= DD;
  int h = tmp % HH;
  int m = tmp / HH;
  const float* W = (m == 0) ? wq : ((m == 1) ? wk : wv);
  wt[i] = (_Float16)W[((size_t)h * CC + c) * DD + d];
}

__global__ void conv_wo_kernel(const float* __restrict__ wo, _Float16* __restrict__ woh, int n) {
  int i = blockIdx.x * blockDim.x + threadIdx.x;
  if (i < n) woh[i] = (_Float16)wo[i];
}

// ---------------------------------------------------------------- QKV projection
// One wave -> 64x64 output macro-tile (4x4 WMMA tiles, 16 accumulators) of one
// (mat, head): full D=64 of head h for q, k, or v.
// q pre-scaled by D^-0.5 = 0.125 ; v stored transposed [B,H,D,T].
__global__ __launch_bounds__(128) void qkv_kernel(const _Float16* __restrict__ xh,
                                                  const _Float16* __restrict__ wt,
                                                  _Float16* __restrict__ qh,
                                                  _Float16* __restrict__ kh,
                                                  _Float16* __restrict__ vth) {
  const int wave = blockIdx.x * 4 + (threadIdx.x >> 5);
  const int lane = threadIdx.x & 31;
  const int ROWB = (BB * TT) / 64;             // 64 macro row blocks
  int rb = wave % ROWB;
  int cb = wave / ROWB;                        // 0..47 : (mat, head)
  int mat = cb / HH;                           // 0=q 1=k 2=v
  int h   = cb % HH;
  int r0  = rb * 64;

  const _Float16* A  = xh + (size_t)r0 * CC;
  const _Float16* Bt = wt + (((size_t)mat * HH + h) * DD) * CC;

  v8f acc[4][4] = {};
  for (int k0 = 0; k0 < CC; k0 += 32) {
    __builtin_prefetch(A + ((lane & 15) * CC) + k0 + 128, 0, 3);
    v16h a[4], bfr[4];
    #pragma unroll
    for (int rt = 0; rt < 4; ++rt) a[rt] = load_a_frag(A + (size_t)rt * 16 * CC, CC, lane, k0);
    #pragma unroll
    for (int ct = 0; ct < 4; ++ct) bfr[ct] = load_b_frag(Bt + (size_t)ct * 16 * CC, CC, lane, k0);
    #pragma unroll
    for (int rt = 0; rt < 4; ++rt)
      #pragma unroll
      for (int ct = 0; ct < 4; ++ct)
        acc[rt][ct] = __builtin_amdgcn_wmma_f32_16x16x32_f16(
            false, a[rt], false, bfr[ct], (short)0, acc[rt][ct], false, false);
  }

  const int n  = lane & 15;
  const int hi = lane >> 4;
  if (mat == 2) {
    #pragma unroll
    for (int rt = 0; rt < 4; ++rt)
      #pragma unroll
      for (int ct = 0; ct < 4; ++ct)
        #pragma unroll
        for (int r = 0; r < 8; ++r) {
          int bt = r0 + rt * 16 + r + 8 * hi;
          int b  = bt >> 11;
          int t  = bt & (TT - 1);
          int d  = ct * 16 + n;
          vth[(((size_t)b * HH + h) * DD + d) * TT + t] = (_Float16)acc[rt][ct][r];
        }
  } else {
    _Float16* dst = (mat == 0) ? qh : kh;
    const float scale = (mat == 0) ? 0.125f : 1.0f;
    #pragma unroll
    for (int rt = 0; rt < 4; ++rt)
      #pragma unroll
      for (int ct = 0; ct < 4; ++ct)
        #pragma unroll
        for (int r = 0; r < 8; ++r) {
          int bt = r0 + rt * 16 + r + 8 * hi;
          int b  = bt >> 11;
          int t  = bt & (TT - 1);
          int d  = ct * 16 + n;
          dst[(((size_t)b * HH + h) * TT + t) * DD + d] = (_Float16)(acc[rt][ct][r] * scale);
        }
  }
}

// ---------------------------------------------------------------- flash attention
// One wave -> 32 query rows (2 row-tiles) of one (b,h); streams 32-wide key
// blocks with online softmax. bk / bv fragments are shared across both
// row-tiles (2x global-load reuse). P goes through LDS (C-layout -> A-layout).
__global__ __launch_bounds__(128) void attn_kernel(const _Float16* __restrict__ qh,
                                                   const _Float16* __restrict__ kh,
                                                   const _Float16* __restrict__ vth,
                                                   _Float16* __restrict__ aoh) {
  __shared__ __align__(16) _Float16 plds[4][2][16 * 32];
  const int wid  = threadIdx.x >> 5;
  const int lane = threadIdx.x & 31;
  const int wave = blockIdx.x * 4 + wid;
  const int QT = TT / 32;                      // 64 macro query blocks
  int qt = wave % QT;
  int bh = wave / QT;                          // 0..B*H-1
  int h = bh % HH, b = bh / HH;
  int q0 = qt * 32;

  const _Float16* Q  = qh  + (size_t)bh * TT * DD;
  const _Float16* K  = kh  + (size_t)bh * TT * DD;
  const _Float16* Vt = vth + (size_t)bh * DD * TT;

  // Q fragments: 2 row-tiles x K-dim D=64 (q already scaled by 1/sqrt(D))
  v16h aq[2][2];
  #pragma unroll
  for (int rt = 0; rt < 2; ++rt) {
    aq[rt][0] = load_a_frag(Q + (size_t)(q0 + rt * 16) * DD, DD, lane, 0);
    aq[rt][1] = load_a_frag(Q + (size_t)(q0 + rt * 16) * DD, DD, lane, 32);
  }

  v8f o[2][4] = {};
  float mrow[2][8], lrow[2][8];
  #pragma unroll
  for (int rt = 0; rt < 2; ++rt)
    #pragma unroll
    for (int r = 0; r < 8; ++r) { mrow[rt][r] = -3.0e30f; lrow[rt][r] = 0.0f; }

  const int n  = lane & 15;
  const int hi = lane >> 4;

  const int nkb = q0 / 32 + 1;                 // causal frontier in 32-blocks
  for (int kb = 0; kb < nkb; ++kb) {
    const int s0 = kb * 32;

    // K^T B-fragments: 2 col-tiles x 2 K-chunks, shared across row-tiles
    v16h bk[2][2];
    #pragma unroll
    for (int ct = 0; ct < 2; ++ct) {
      bk[ct][0] = load_b_frag(K + (size_t)(s0 + ct * 16) * DD, DD, lane, 0);
      bk[ct][1] = load_b_frag(K + (size_t)(s0 + ct * 16) * DD, DD, lane, 32);
    }

    // S = q k^T : 2x2 tiles of 16x16, K=64 split 32+32
    v8f S[2][2] = {};
    #pragma unroll
    for (int rt = 0; rt < 2; ++rt)
      #pragma unroll
      for (int ct = 0; ct < 2; ++ct) {
        S[rt][ct] = __builtin_amdgcn_wmma_f32_16x16x32_f16(
            false, aq[rt][0], false, bk[ct][0], (short)0, S[rt][ct], false, false);
        S[rt][ct] = __builtin_amdgcn_wmma_f32_16x16x32_f16(
            false, aq[rt][1], false, bk[ct][1], (short)0, S[rt][ct], false, false);
      }

    // causal mask + online softmax per row-tile; P into LDS
    #pragma unroll
    for (int rt = 0; rt < 2; ++rt) {
      _Float16* pl = plds[wid][rt];
      float fac[8];
      #pragma unroll
      for (int r = 0; r < 8; ++r) {
        int row = q0 + rt * 16 + r + 8 * hi;
        float x0 = ((s0 + n)      <= row) ? S[rt][0][r] : -3.0e30f;
        float x1 = ((s0 + 16 + n) <= row) ? S[rt][1][r] : -3.0e30f;
        float mx = fmaxf(x0, x1);
        #pragma unroll
        for (int msk = 1; msk < 16; msk <<= 1) mx = fmaxf(mx, __shfl_xor(mx, msk, 32));
        float mnew = fmaxf(mrow[rt][r], mx);
        float f  = __expf(mrow[rt][r] - mnew);
        float p0 = __expf(x0 - mnew);
        float p1 = __expf(x1 - mnew);
        float sum = p0 + p1;
        #pragma unroll
        for (int msk = 1; msk < 16; msk <<= 1) sum += __shfl_xor(sum, msk, 32);
        lrow[rt][r] = lrow[rt][r] * f + sum;
        mrow[rt][r] = mnew;
        fac[r] = f;
        pl[(r + 8 * hi) * 32 + n]      = (_Float16)p0;
        pl[(r + 8 * hi) * 32 + 16 + n] = (_Float16)p1;
      }
      #pragma unroll
      for (int dt = 0; dt < 4; ++dt)
        #pragma unroll
        for (int r = 0; r < 8; ++r) o[rt][dt][r] *= fac[r];
    }

    asm volatile("s_wait_dscnt 0" ::: "memory");    // P tiles visible for re-read

    // V B-fragments (transposed V: contiguous K), shared across row-tiles
    v16h bv[4];
    #pragma unroll
    for (int dt = 0; dt < 4; ++dt)
      bv[dt] = load_b_frag(Vt + (size_t)(dt * 16) * TT + s0, TT, lane, 0);

    #pragma unroll
    for (int rt = 0; rt < 2; ++rt) {
      v16h ap = load_a_frag(plds[wid][rt], 32, lane, 0);
      #pragma unroll
      for (int dt = 0; dt < 4; ++dt)
        o[rt][dt] = __builtin_amdgcn_wmma_f32_16x16x32_f16(
            false, ap, false, bv[dt], (short)0, o[rt][dt], false, false);
    }
  }

  // normalize + store concat-head layout [B,T,C], C slot = h*64 + d
  #pragma unroll
  for (int rt = 0; rt < 2; ++rt)
    #pragma unroll
    for (int r = 0; r < 8; ++r) {
      float inv = 1.0f / lrow[rt][r];
      int t = q0 + rt * 16 + r + 8 * hi;
      size_t rowbase = ((size_t)b * TT + t) * CC + h * DD;
      #pragma unroll
      for (int dt = 0; dt < 4; ++dt)
        aoh[rowbase + dt * 16 + n] = (_Float16)(o[rt][dt][r] * inv);
    }
}

// ---------------------------------------------------------------- output projection
// y = attn_out @ Wo^T + bo ; 64x64 macro-tile per wave (16 accumulators).
__global__ __launch_bounds__(128) void oproj_kernel(const _Float16* __restrict__ aoh,
                                                    const _Float16* __restrict__ woh,
                                                    const float* __restrict__ bo,
                                                    float* __restrict__ out) {
  const int wave = blockIdx.x * 4 + (threadIdx.x >> 5);
  const int lane = threadIdx.x & 31;
  const int ROWB = (BB * TT) / 64;             // 64
  int rb = wave % ROWB;
  int cb = wave / ROWB;                        // 0..15
  int r0 = rb * 64;
  int c0 = cb * 64;

  const _Float16* A  = aoh + (size_t)r0 * CC;
  const _Float16* Bt = woh + (size_t)c0 * CC;

  v8f acc[4][4] = {};
  for (int k0 = 0; k0 < CC; k0 += 32) {
    v16h a[4], bfr[4];
    #pragma unroll
    for (int rt = 0; rt < 4; ++rt) a[rt] = load_a_frag(A + (size_t)rt * 16 * CC, CC, lane, k0);
    #pragma unroll
    for (int ct = 0; ct < 4; ++ct) bfr[ct] = load_b_frag(Bt + (size_t)ct * 16 * CC, CC, lane, k0);
    #pragma unroll
    for (int rt = 0; rt < 4; ++rt)
      #pragma unroll
      for (int ct = 0; ct < 4; ++ct)
        acc[rt][ct] = __builtin_amdgcn_wmma_f32_16x16x32_f16(
            false, a[rt], false, bfr[ct], (short)0, acc[rt][ct], false, false);
  }

  const int n  = lane & 15;
  const int hi = lane >> 4;
  #pragma unroll
  for (int ct = 0; ct < 4; ++ct) {
    float bias = bo[c0 + ct * 16 + n];
    #pragma unroll
    for (int rt = 0; rt < 4; ++rt)
      #pragma unroll
      for (int r = 0; r < 8; ++r) {
        int row = r0 + rt * 16 + r + 8 * hi;
        out[(size_t)row * CC + c0 + ct * 16 + n] = acc[rt][ct][r] + bias;
      }
  }
}

// ---------------------------------------------------------------- launcher

extern "C" void kernel_launch(void* const* d_in, const int* in_sizes, int n_in,
                              void* d_out, int out_size, void* d_ws, size_t ws_size,
                              hipStream_t stream) {
  const float* x  = (const float*)d_in[0];
  const float* Wq = (const float*)d_in[1];
  const float* Wk = (const float*)d_in[2];
  const float* Wv = (const float*)d_in[3];
  const float* Wo = (const float*)d_in[4];
  const float* bo = (const float*)d_in[5];
  float* out = (float*)d_out;

  char* ws = (char*)d_ws;
  const size_t SZ_XH = (size_t)BB * TT * CC * 2;       //  8 MB
  const size_t SZ_WT = (size_t)3 * HH * DD * CC * 2;   //  6 MB
  const size_t SZ_WO = (size_t)CC * CC * 2;            //  2 MB
  const size_t SZ_Q  = (size_t)BB * HH * TT * DD * 2;  //  8 MB
  _Float16* xh  = (_Float16*)(ws);
  _Float16* wt  = (_Float16*)(ws + SZ_XH);
  _Float16* woh = (_Float16*)(ws + SZ_XH + SZ_WT);
  _Float16* qh  = (_Float16*)(ws + SZ_XH + SZ_WT + SZ_WO);
  _Float16* kh  = (_Float16*)(ws + SZ_XH + SZ_WT + SZ_WO + SZ_Q);
  _Float16* vth = (_Float16*)(ws + SZ_XH + SZ_WT + SZ_WO + 2 * SZ_Q);
  _Float16* aoh = (_Float16*)(ws + SZ_XH + SZ_WT + SZ_WO + 3 * SZ_Q);

  // 1) fp32 -> fp16 staging (+ weight transposes)
  {
    int n = BB * TT * CC;
    conv_x_kernel<<<(n + 255) / 256, 256, 0, stream>>>(x, xh, n);
  }
  {
    int n = 3 * HH * DD * CC;
    conv_w_kernel<<<(n + 255) / 256, 256, 0, stream>>>(Wq, Wk, Wv, wt);
  }
  {
    int n = CC * CC;
    conv_wo_kernel<<<(n + 255) / 256, 256, 0, stream>>>(Wo, woh, n);
  }

  // 2) QKV projection: 64 row-blocks x 48 (mat,head) = 3072 waves, 4/block
  qkv_kernel<<<3072 / 4, 128, 0, stream>>>(xh, wt, qh, kh, vth);

  // 3) flash attention: B*H*(T/32) = 2048 waves
  attn_kernel<<<2048 / 4, 128, 0, stream>>>(qh, kh, vth, aoh);

  // 4) output projection: 64 x 16 = 1024 waves
  oproj_kernel<<<1024 / 4, 128, 0, stream>>>(aoh, woh, bo, out);
}
